// PatchEmbedding_88055419503373
// MI455X (gfx1250) — matrix operand order repaired and checked by reading, not compiled, not run on hardware
//
#include <hip/hip_runtime.h>

typedef __attribute__((ext_vector_type(16))) _Float16 v16h;
typedef __attribute__((ext_vector_type(8)))  float    v8f;
typedef __attribute__((ext_vector_type(4)))  float    f32x4;

#define D_DIM   768
#define PATCHES 196          // 14*14 patches per image
#define ROWSTR  672          // 224*3 floats per image row
#define M_DIM   (64 * PATCHES)   // 12544
#define KTILES  (D_DIM / 32)     // 24
#define MTILES  (M_DIM / 16)     // 784
#define NTILES  (D_DIM / 16)     // 48

// ---------------------------------------------------------------------------
// Branch-free exact GELU: 0.5*x*(1+erf(x/sqrt(2))).
// erf via Abramowitz-Stegun 7.1.26 (|err| <= 1.5e-7): straight-line VALU +
// one v_rcp_f32 + one v_exp_f32 (TRANS32 ops co-execute with VALU on CDNA5),
// no EXEC-mask divergence (unlike OCML erff).
// ---------------------------------------------------------------------------
__device__ __forceinline__ float gelu_exact(float x) {
    float z  = x * 0.70710678118654752f;
    float az = __builtin_fabsf(z);
    float t  = __builtin_amdgcn_rcpf(__builtin_fmaf(0.3275911f, az, 1.0f));
    float p  = __builtin_fmaf(t, 1.061405429f, -1.453152027f);
    p = __builtin_fmaf(t, p, 1.421413741f);
    p = __builtin_fmaf(t, p, -0.284496736f);
    p = __builtin_fmaf(t, p, 0.254829592f);
    p = p * t;
    float e  = __expf(-az * az);                 // v_exp_f32 path
    float er = __builtin_fmaf(-p, e, 1.0f);      // erf(|z|)
    er = __builtin_copysignf(er, z);
    return 0.5f * x * (1.0f + er);
}

// ===========================================================================
// Pre-pass 1: im2col the image into A fragments, f16, WMMA-fragment order.
// Layout: tile (mt, kt) -> 32 lanes -> 16 halves (32 B) per lane.
//   lane<16 : row = mt*16+lane,    K = {kt*32+0..7,   kt*32+8..15}
//   lane>=16: row = mt*16+lane-16, K = {kt*32+16..23, kt*32+24..31}
// Each 8-half K-group is a contiguous 16B-aligned run of 8 floats in the
// image (48 = 16*3 floats per patch pixel-row is a multiple of 8).
// ===========================================================================
__global__ __launch_bounds__(256) void a_pack_kernel(const float* __restrict__ img,
                                                     _Float16* __restrict__ Apk) {
    int idx  = blockIdx.x * 256 + threadIdx.x;   // MTILES*KTILES*32 = 602112 exact
    int lane = idx & 31;
    int tile = idx >> 5;
    int kt   = tile % KTILES;
    int mt   = tile / KTILES;

    int row = mt * 16 + (lane & 15);
    int g1  = kt * 32 + (lane >> 4) * 16;
    int g2  = g1 + 8;

    int b   = row / PATCHES;
    int rem = row - b * PATCHES;
    int pi  = rem / 14;
    int pj  = rem - pi * 14;
    int base = ((b * 224 + pi * 16) * 224 + pj * 16) * 3;

    int r1 = g1 / 48, o1 = g1 - r1 * 48;
    int r2 = g2 / 48, o2 = g2 - r2 * 48;
    const float* p1 = img + base + r1 * ROWSTR + o1;
    const float* p2 = img + base + r2 * ROWSTR + o2;

    f32x4 x0 = *(const f32x4*)p1;
    f32x4 x1 = *(const f32x4*)(p1 + 4);
    f32x4 x2 = *(const f32x4*)p2;
    f32x4 x3 = *(const f32x4*)(p2 + 4);

    v16h t;
    #pragma unroll
    for (int e = 0; e < 4; ++e) {
        t[e]      = (_Float16)x0[e];
        t[4 + e]  = (_Float16)x1[e];
        t[8 + e]  = (_Float16)x2[e];
        t[12 + e] = (_Float16)x3[e];
    }
    ((v16h*)Apk)[idx] = t;                       // 32B coalesced store
}

// ===========================================================================
// Pre-pass 2: pack W[K=768][N=768] f32 into B fragments, f16, fragment order.
// Layout: tile (nt, kt) -> 32 lanes -> 16 halves per lane.
//   lane<16 : col = nt*16+lane,    K = kt*32 + 0..15
//   lane>=16: col = nt*16+lane-16, K = kt*32 + 16..31
// ===========================================================================
__global__ __launch_bounds__(256) void b_pack_kernel(const float* __restrict__ W,
                                                     _Float16* __restrict__ Bpk) {
    int idx  = blockIdx.x * 256 + threadIdx.x;   // NTILES*KTILES*32 = 36864 exact
    int lane = idx & 31;
    int tile = idx >> 5;
    int kt   = tile % KTILES;
    int nt   = tile / KTILES;

    int col = nt * 16 + (lane & 15);
    int k0  = kt * 32 + (lane >> 4) * 16;

    v16h t;
    #pragma unroll
    for (int e = 0; e < 16; ++e)
        t[e] = (_Float16)W[(k0 + e) * D_DIM + col];  // coalesced across 16 lanes
    ((v16h*)Bpk)[idx] = t;
}

// ===========================================================================
// Fast GEMM: C = A x B, bias + exact GELU epilogue.
// Block 256 (8 wave32) = 2(M) x 4(N) waves; block tile 128 x 256.
// Wave tile 64 x 64 = 4x4 wmma_f32_16x16x32_f16; 16 WMMAs per 8 v16h loads.
// ===========================================================================
__global__ __launch_bounds__(256) void gemm_packed_kernel(
    const _Float16* __restrict__ Apk, const _Float16* __restrict__ Bpk,
    const float* __restrict__ bias, float* __restrict__ out)
{
    const int lane   = threadIdx.x & 31;
    const int wid    = threadIdx.x >> 5;
    const int wave_m = wid & 1;                  // 0..1
    const int wave_n = wid >> 1;                 // 0..3
    const int m0 = blockIdx.y * 128 + wave_m * 64;
    const int n0 = blockIdx.x * 256 + wave_n * 64;
    const int lhalf = lane >> 4;
    const int l15   = lane & 15;

    const v16h* Ap = (const v16h*)Apk + (size_t)(m0 >> 4) * (KTILES * 32) + lane;
    const v16h* Bp = (const v16h*)Bpk + (size_t)(n0 >> 4) * (KTILES * 32) + lane;

    v8f acc[4][4];
    #pragma unroll
    for (int i = 0; i < 4; ++i)
        #pragma unroll
        for (int j = 0; j < 4; ++j)
            acc[i][j] = (v8f){0.f, 0.f, 0.f, 0.f, 0.f, 0.f, 0.f, 0.f};

    for (int kk = 0; kk < KTILES; ++kk) {
        v16h a[4], b[4];
        #pragma unroll
        for (int i = 0; i < 4; ++i) a[i] = Ap[i * (KTILES * 32) + kk * 32];
        #pragma unroll
        for (int j = 0; j < 4; ++j) b[j] = Bp[j * (KTILES * 32) + kk * 32];

        #pragma unroll
        for (int i = 0; i < 4; ++i)
            #pragma unroll
            for (int j = 0; j < 4; ++j)
                acc[i][j] = __builtin_amdgcn_wmma_f32_16x16x32_f16(
                    false, a[i], false, b[j], (short)0, acc[i][j], false, false);
    }

    // C/D layout: element v in lane L -> M = v + 8*(L>=16), N = L&15.
    #pragma unroll
    for (int j = 0; j < 4; ++j) {
        const int col = n0 + j * 16 + l15;
        const float bv = bias[col];
        #pragma unroll
        for (int i = 0; i < 4; ++i) {
            const int rbase = m0 + i * 16 + lhalf * 8;
            v8f c = acc[i][j];
            #pragma unroll
            for (int v = 0; v < 8; ++v)
                out[(size_t)(rbase + v) * D_DIM + col] = gelu_exact(c[v] + bv);
        }
    }
}

// ===========================================================================
// Fallback path (needs only 1.18 MB workspace): Wt[N][K] f16 pre-pass +
// fused on-the-fly im2col GEMM (32x64 wave tile).
// ===========================================================================
__global__ __launch_bounds__(256) void wt_pack_kernel(const float* __restrict__ W,
                                                      _Float16* __restrict__ Wt) {
    int idx = blockIdx.x * 256 + threadIdx.x;
    int k = idx / D_DIM;
    int n = idx - k * D_DIM;
    Wt[n * D_DIM + k] = (_Float16)W[idx];
}

__global__ __launch_bounds__(256) void gemm_fallback_kernel(
    const float* __restrict__ img, const _Float16* __restrict__ Wt,
    const float* __restrict__ bias, float* __restrict__ out)
{
    const int lane   = threadIdx.x & 31;
    const int wid    = threadIdx.x >> 5;
    const int wave_m = wid & 3;
    const int wave_n = wid >> 2;
    const int m0 = blockIdx.y * 128 + wave_m * 32;
    const int n0 = blockIdx.x * 128 + wave_n * 64;
    const int lhalf = lane >> 4;
    const int l15   = lane & 15;

    int fofs[2][2];
    {
        const int g0 = lhalf * 16;
        #pragma unroll
        for (int i = 0; i < 2; ++i) {
            int row = m0 + i * 16 + l15;
            int b   = row / PATCHES;
            int rem = row - b * PATCHES;
            int pi  = rem / 14;
            int pj  = rem - pi * 14;
            int base = ((b * 224 + pi * 16) * 224 + pj * 16) * 3;
            fofs[i][0] = base + g0;
            fofs[i][1] = base + g0 + 8;
        }
    }
    int off1 = lhalf * 16;
    int off2 = off1 + 8;

    v8f acc[2][4];
    #pragma unroll
    for (int i = 0; i < 2; ++i)
        #pragma unroll
        for (int j = 0; j < 4; ++j)
            acc[i][j] = (v8f){0.f, 0.f, 0.f, 0.f, 0.f, 0.f, 0.f, 0.f};

    #pragma unroll 2
    for (int kb = 0; kb < D_DIM; kb += 32) {
        v16h a[2];
        #pragma unroll
        for (int i = 0; i < 2; ++i) {
            f32x4 x0 = *(const f32x4*)(img + fofs[i][0]);
            f32x4 x1 = *(const f32x4*)(img + fofs[i][0] + 4);
            f32x4 x2 = *(const f32x4*)(img + fofs[i][1]);
            f32x4 x3 = *(const f32x4*)(img + fofs[i][1] + 4);
            v16h t;
            #pragma unroll
            for (int e = 0; e < 4; ++e) {
                t[e]      = (_Float16)x0[e];
                t[4 + e]  = (_Float16)x1[e];
                t[8 + e]  = (_Float16)x2[e];
                t[12 + e] = (_Float16)x3[e];
            }
            a[i] = t;
        }

        const int koff = kb + lhalf * 16;
        v16h b[4];
        #pragma unroll
        for (int j = 0; j < 4; ++j) {
            int n = n0 + j * 16 + l15;
            b[j] = *(const v16h*)(Wt + n * D_DIM + koff);
        }

        #pragma unroll
        for (int i = 0; i < 2; ++i)
            #pragma unroll
            for (int j = 0; j < 4; ++j)
                acc[i][j] = __builtin_amdgcn_wmma_f32_16x16x32_f16(
                    false, a[i], false, b[j], (short)0, acc[i][j], false, false);

        bool w1 = (off1 + 32) >= 48;
        bool w2 = (off2 + 32) >= 48;
        off1 += 32 - (w1 ? 48 : 0);
        off2 += 32 - (w2 ? 48 : 0);
        int d1 = w1 ? (32 + ROWSTR - 48) : 32;
        int d2 = w2 ? (32 + ROWSTR - 48) : 32;
        fofs[0][0] += d1; fofs[1][0] += d1;
        fofs[0][1] += d2; fofs[1][1] += d2;
    }

    #pragma unroll
    for (int j = 0; j < 4; ++j) {
        const int col = n0 + j * 16 + l15;
        const float bv = bias[col];
        #pragma unroll
        for (int i = 0; i < 2; ++i) {
            const int rbase = m0 + i * 16 + lhalf * 8;
            v8f c = acc[i][j];
            #pragma unroll
            for (int v = 0; v < 8; ++v)
                out[(size_t)(rbase + v) * D_DIM + col] = gelu_exact(c[v] + bv);
        }
    }
}

// ===========================================================================
extern "C" void kernel_launch(void* const* d_in, const int* in_sizes, int n_in,
                              void* d_out, int out_size, void* d_ws, size_t ws_size,
                              hipStream_t stream) {
    const float* image = (const float*)d_in[0];   // [64,224,224,3] f32
    const float* Wp    = (const float*)d_in[1];   // [768,768] f32
    const float* bp    = (const float*)d_in[2];   // [768] f32
    float* out         = (float*)d_out;           // [64,196,768] f32

    const size_t sizeA = (size_t)M_DIM * D_DIM * sizeof(_Float16);  // 19.27 MB
    const size_t sizeB = (size_t)D_DIM * D_DIM * sizeof(_Float16);  //  1.18 MB

    if (ws_size >= sizeA + sizeB) {
        _Float16* Apk = (_Float16*)d_ws;
        _Float16* Bpk = (_Float16*)((char*)d_ws + sizeA);

        a_pack_kernel<<<(MTILES * KTILES * 32) / 256, 256, 0, stream>>>(image, Apk);
        b_pack_kernel<<<(NTILES * KTILES * 32) / 256, 256, 0, stream>>>(Wp, Bpk);

        dim3 grid(D_DIM / 256 /*=3*/, M_DIM / 128 /*=98*/);
        gemm_packed_kernel<<<grid, 256, 0, stream>>>(Apk, Bpk, bp, out);
    } else {
        _Float16* Wt = (_Float16*)d_ws;           // needs 1.18 MB only
        wt_pack_kernel<<<(D_DIM * D_DIM) / 256, 256, 0, stream>>>(Wp, Wt);
        dim3 grid(D_DIM / 128 /*=6*/, M_DIM / 128 /*=98*/);
        gemm_fallback_kernel<<<grid, 256, 0, stream>>>(image, Wt, bp, out);
    }
}
